// TopKSparseAutoencoder_82583631167932
// MI455X (gfx1250) — matrix operand order repaired and checked by reading, not compiled, not run on hardware
//
#include <hip/hip_runtime.h>
#include <hip/hip_bf16.h>

// ---------------- problem constants ----------------
constexpr int BB = 16384;   // batch
constexpr int DD = 768;     // model dim
constexpr int FF = 8192;    // features
constexpr int KK = 32;      // top-k

// ---------------- vector types ----------------
typedef __attribute__((ext_vector_type(16))) __bf16 v16bf;
typedef __attribute__((ext_vector_type(8)))  float  v8f;
typedef __attribute__((ext_vector_type(8)))  __bf16 bf16x8;

// ============================================================
// Kernel P0: x_c = bf16(x - dec_bias)   [B][D] row-major
// ============================================================
__global__ __launch_bounds__(256)
void prep_x(const float* __restrict__ x, const float* __restrict__ db,
            __bf16* __restrict__ xc) {
    const int b   = blockIdx.x;
    const int tid = threadIdx.x;
    #pragma unroll
    for (int j = 0; j < DD / 256; ++j) {
        const int d = tid + j * 256;
        xc[(size_t)b * DD + d] = (__bf16)(x[(size_t)b * DD + d] - db[d]);
    }
}

// ============================================================
// Kernel P1: W_encT = bf16(W_enc^T)   [D][F] (K-major for WMMA B tiles)
// ============================================================
__global__ __launch_bounds__(256)
void prep_wenc(const float* __restrict__ We, __bf16* __restrict__ wT) {
    __shared__ float tile[32][33];
    const int tx = threadIdx.x;          // 0..31
    const int ty = threadIdx.y;          // 0..7
    const int f0 = blockIdx.x * 32;
    const int d0 = blockIdx.y * 32;
    #pragma unroll
    for (int i = 0; i < 4; ++i)
        tile[ty + i * 8][tx] = We[(size_t)(f0 + ty + i * 8) * DD + d0 + tx];
    __syncthreads();
    #pragma unroll
    for (int i = 0; i < 4; ++i)
        wT[(size_t)(d0 + ty + i * 8) * FF + f0 + tx] = (__bf16)tile[tx][ty + i * 8];
}

// ============================================================
// Kernel P2: transpose W_dec [D][F] -> W_decT [F][D] f32 (coalesced decode)
// ============================================================
__global__ __launch_bounds__(256)
void transpose_wdec(const float* __restrict__ Wd, float* __restrict__ WdT) {
    __shared__ float tile[32][33];
    const int tx = threadIdx.x;
    const int ty = threadIdx.y;
    const int f0 = blockIdx.x * 32;
    const int d0 = blockIdx.y * 32;
    #pragma unroll
    for (int i = 0; i < 4; ++i)
        tile[ty + i * 8][tx] = Wd[(size_t)(d0 + ty + i * 8) * FF + f0 + tx];
    __syncthreads();
    #pragma unroll
    for (int i = 0; i < 4; ++i)
        WdT[(size_t)(f0 + ty + i * 8) * DD + d0 + tx] = tile[tx][ty + i * 8];
}

// ============================================================
// Kernel 1: encoder GEMM  acts = relu(x_c @ W_enc^T + b_enc)
//   block tile M=64 x N=256, K-step 32; 8 waves: wave(mg 0..3, ng 0..1) owns 16x128
//   B tile: async global->LDS (double buffered, ASYNCcnt pipeline)
//   A frags: direct bf16 loads from L2-resident x_c
// ============================================================
constexpr int TM = 64;
constexpr int TN = 256;
constexpr int TK = 32;
constexpr int BS_STR = 264;   // bf16 elems per LDS B row: 528B (16B aligned, conflict pad)
constexpr int NSTEP = DD / TK;

__global__ __launch_bounds__(256)
void enc_gemm(const __bf16* __restrict__ xc,
              const __bf16* __restrict__ wT,
              const float* __restrict__ benc,
              float* __restrict__ acts) {
    __shared__ __bf16 BsT[2][TK][BS_STR];   // (k, n) bf16, double buffered

    const int tid   = threadIdx.x;
    const int nbase = blockIdx.x * TN;
    const int mbase = blockIdx.y * TM;
    const int wave  = tid >> 5;
    const int lane  = tid & 31;
    const int mg    = wave >> 1;    // 0..3
    const int ng    = wave & 1;     // 0..1

    // ---- async B-tile staging: 1024 x 16B transfers, 4 per thread ----
    const int bk  = tid >> 5;          // base row contribution per j handled below
    (void)bk;

    auto issueB = [&](int buf, int d0) {
        #pragma unroll
        for (int j = 0; j < 4; ++j) {
            const int tr  = j * 256 + tid;      // 0..1023
            const int k   = tr >> 5;            // 0..31
            const int seg = tr & 31;            // 0..31 (16B segments)
            const __bf16* g = wT + (size_t)(d0 + k) * FF + nbase + seg * 8;
            const unsigned lds = (unsigned)(uintptr_t)&BsT[buf][k][seg * 8];
            asm volatile("global_load_async_to_lds_b128 %0, %1, off"
                         :: "v"(lds), "v"((unsigned long long)(uintptr_t)g)
                         : "memory");
        }
    };

    v8f acc[8] = {};

    // A fragment source: lane l<16 holds K 0-7 & 16-23; l>=16 holds K 8-15 & 24-31
    const int arow = mbase + mg * 16 + (lane & 15);
    const int ksel = (lane >> 4) * 8;
    const __bf16* aptr = xc + (size_t)arow * DD + ksel;

    issueB(0, 0);

    for (int step = 0; step < NSTEP; ++step) {
        const int d0  = step * TK;
        const int buf = step & 1;

        if (step + 1 < NSTEP) {
            issueB(buf ^ 1, d0 + TK);
            asm volatile("s_wait_asynccnt 4" ::: "memory");   // current buffer done
        } else {
            asm volatile("s_wait_asynccnt 0" ::: "memory");
        }
        __syncthreads();

        const bf16x8 alo = *(const bf16x8*)(aptr + d0);
        const bf16x8 ahi = *(const bf16x8*)(aptr + d0 + 16);
        const v16bf a = __builtin_shufflevector(alo, ahi,
            0, 1, 2, 3, 4, 5, 6, 7, 8, 9, 10, 11, 12, 13, 14, 15);

        #pragma unroll
        for (int t = 0; t < 8; ++t) {
            const __bf16* bp = &BsT[buf][(lane >> 4) * 8][ng * 128 + t * 16 + (lane & 15)];
            v16bf b;
            #pragma unroll
            for (int s = 0; s < 8; ++s) {
                b[s]     = bp[s * BS_STR];
                b[8 + s] = bp[(16 + s) * BS_STR];
            }
            acc[t] = __builtin_amdgcn_wmma_f32_16x16x32_bf16(
                false, a, false, b, (short)0, acc[t], false, false);
        }
        __syncthreads();
    }

    // ---- epilogue: + b_enc, relu, store f32 ----
    const int rbase = mbase + mg * 16 + ((lane >> 4) * 8);
    #pragma unroll
    for (int t = 0; t < 8; ++t) {
        const int col = nbase + ng * 128 + t * 16 + (lane & 15);
        const float be = benc[col];
        #pragma unroll
        for (int r = 0; r < 8; ++r) {
            const float v = acc[t][r] + be;
            acts[(size_t)(rbase + r) * FF + col] = v > 0.f ? v : 0.f;
        }
    }
}

// ============================================================
// Kernel 2: per-row top-K selection -> vals/idx in workspace
// ============================================================
__global__ __launch_bounds__(256)
void select_topk(const float* __restrict__ acts,
                 float* __restrict__ vals, int* __restrict__ sidx) {
    __shared__ float work[FF];
    __shared__ float rv[256];
    __shared__ int   ri[256];
    const int b   = blockIdx.x;
    const int tid = threadIdx.x;

    for (int j = 0; j < FF / 256; ++j)
        work[tid + 256 * j] = acts[(size_t)b * FF + tid + 256 * j];
    __syncthreads();

    for (int k = 0; k < KK; ++k) {
        float bv = -3.402823466e+38f;
        int   bi = 0;
        #pragma unroll 4
        for (int j = 0; j < FF / 256; ++j) {
            const int   i = tid + 256 * j;
            const float v = work[i];
            if (v > bv) { bv = v; bi = i; }
        }
        rv[tid] = bv; ri[tid] = bi;
        __syncthreads();
        for (int s = 128; s > 0; s >>= 1) {
            if (tid < s) {
                const float ov = rv[tid + s];
                const int   oi = ri[tid + s];
                if (ov > rv[tid] || (ov == rv[tid] && oi < ri[tid])) {
                    rv[tid] = ov; ri[tid] = oi;
                }
            }
            __syncthreads();
        }
        if (tid == 0) {
            vals[(size_t)b * KK + k] = rv[0];
            sidx[(size_t)b * KK + k] = ri[0];
            work[ri[0]] = -3.402823466e+38f;
        }
        __syncthreads();
    }
}

// ============================================================
// Kernel 3: zero-fill sparse_code region (float4 per thread)
// ============================================================
__global__ __launch_bounds__(256)
void zero_fill(float4* __restrict__ out4) {
    const size_t i = (size_t)blockIdx.x * 256 + threadIdx.x;
    out4[i] = make_float4(0.f, 0.f, 0.f, 0.f);
}

// ============================================================
// Kernel 4: scatter top-K values into sparse_code
// ============================================================
__global__ __launch_bounds__(256)
void scatter_topk(const float* __restrict__ vals, const int* __restrict__ sidx,
                  float* __restrict__ sparse) {
    const size_t t = (size_t)blockIdx.x * 256 + threadIdx.x;   // t < B*K
    const size_t b = t / KK;
    sparse[b * FF + (size_t)sidx[t]] = vals[t];
}

// ============================================================
// Kernel 5: sparse decode  recon[b,:] = dec_bias + sum_k val*W_decT[f,:]
// ============================================================
__global__ __launch_bounds__(256)
void decode(const float* __restrict__ vals, const int* __restrict__ sidx,
            const float* __restrict__ WdT, const float* __restrict__ dbias,
            float* __restrict__ recon) {
    __shared__ float sv[KK];
    __shared__ int   si[KK];
    const int b   = blockIdx.x;
    const int tid = threadIdx.x;
    if (tid < KK) {
        sv[tid] = vals[(size_t)b * KK + tid];
        si[tid] = sidx[(size_t)b * KK + tid];
    }
    __syncthreads();
    float a0 = dbias[tid], a1 = dbias[256 + tid], a2 = dbias[512 + tid];
    #pragma unroll 4
    for (int k = 0; k < KK; ++k) {
        const float v = sv[k];
        const float* col = WdT + (size_t)si[k] * DD;
        a0 += v * col[tid];
        a1 += v * col[256 + tid];
        a2 += v * col[512 + tid];
    }
    float* r = recon + (size_t)b * DD;
    r[tid] = a0; r[256 + tid] = a1; r[512 + tid] = a2;
}

// ============================================================
// launcher
// ============================================================
extern "C" void kernel_launch(void* const* d_in, const int* in_sizes, int n_in,
                              void* d_out, int out_size, void* d_ws, size_t ws_size,
                              hipStream_t stream) {
    const float* x     = (const float*)d_in[0];
    const float* Wenc  = (const float*)d_in[1];
    const float* benc  = (const float*)d_in[2];
    const float* Wdec  = (const float*)d_in[3];
    const float* dbias = (const float*)d_in[4];

    float* recon  = (float*)d_out;                       // [B, D]
    float* sparse = (float*)d_out + (size_t)BB * DD;     // [B, F]

    // workspace layout
    float*  WdT  = (float*)d_ws;                         // [F, D] f32
    float*  vals = WdT + (size_t)FF * DD;                // [B, K] f32
    int*    sidx = (int*)(vals + (size_t)BB * KK);       // [B, K] i32
    __bf16* xc   = (__bf16*)(sidx + (size_t)BB * KK);    // [B, D] bf16 (centered)
    __bf16* wT   = xc + (size_t)BB * DD;                 // [D, F] bf16 (W_enc^T)

    // precision / layout pre-passes (all inputs are L2-resident afterwards)
    prep_x<<<BB, 256, 0, stream>>>(x, dbias, xc);
    prep_wenc<<<dim3(FF / 32, DD / 32), dim3(32, 8), 0, stream>>>(Wenc, wT);
    transpose_wdec<<<dim3(FF / 32, DD / 32), dim3(32, 8), 0, stream>>>(Wdec, WdT);

    // encoder GEMM (WMMA bf16, async-LDS double-buffered B tiles)
    enc_gemm<<<dim3(FF / TN, BB / TM), 256, 0, stream>>>(xc, wT, benc, sparse);

    // per-row top-K
    select_topk<<<BB, 256, 0, stream>>>(sparse, vals, sidx);

    // rebuild sparse_code = zeros + scattered top-K
    zero_fill<<<(unsigned)((size_t)BB * FF / 1024), 256, 0, stream>>>((float4*)sparse);
    scatter_topk<<<(unsigned)((size_t)BB * KK / 256), 256, 0, stream>>>(vals, sidx, sparse);

    // sparse decode
    decode<<<BB, 256, 0, stream>>>(vals, sidx, WdT, dbias, recon);
}
